// RPNHead_6459630814147
// MI455X (gfx1250) — matrix operand order repaired and checked by reading, not compiled, not run on hardware
//
#include <hip/hip_runtime.h>
#include <hip/hip_bf16.h>
#include <stdint.h>

#define C_INC 256
#define NCLS  15
#define HH    200
#define WWD   304
#define HWP   (HH*WWD)      // 60800
#define NOUTC 75
#define NPADC 80

typedef __bf16 bf16;
typedef __bf16 v16bf __attribute__((ext_vector_type(16)));
typedef __bf16 v8bf  __attribute__((ext_vector_type(8)));
typedef float  v8f   __attribute__((ext_vector_type(8)));
typedef uint32_t u32x4 __attribute__((ext_vector_type(4)));
typedef uint32_t u32x8 __attribute__((ext_vector_type(8)));

// workspace layout (bytes), all 256-aligned
#define OFF_FT   0ull          // bf16 fT  [HW][256]   31,129,600 B
#define OFF_F0   31129600ull   // bf16 f0  [HW][256]   31,129,600 B
#define OFF_WT   62259200ull   // bf16 wtap[9][256][256]  1,179,648 B
#define OFF_B2   63438848ull   // bf16 B2  [80][256]       40,960 B
#define OFF_BIAS 63479808ull   // f32 bias2[80]               320 B

#define BROW 264               // LDS B row stride in bf16 (512B data + 16B TDM pad)

static __device__ __forceinline__ void wait_async0() {
#if __has_builtin(__builtin_amdgcn_s_wait_asynccnt)
    __builtin_amdgcn_s_wait_asynccnt(0);
#else
    asm volatile("s_wait_asynccnt 0x0" ::: "memory");
#endif
}

static __device__ __forceinline__ void wait_tensor0() {
#if __has_builtin(__builtin_amdgcn_s_wait_tensorcnt)
    __builtin_amdgcn_s_wait_tensorcnt(0);
#else
    asm volatile("s_wait_tensorcnt 0x0" ::: "memory");
#endif
}

static __device__ __forceinline__ void wait_tensor1() {
#if __has_builtin(__builtin_amdgcn_s_wait_tensorcnt)
    __builtin_amdgcn_s_wait_tensorcnt(1);
#else
    asm volatile("s_wait_tensorcnt 0x1" ::: "memory");
#endif
}

// ---------------------------------------------------------------------------
// Transpose NCHW f32 feature -> channels-last bf16 fT[HW][256]
__global__ __launch_bounds__(256)
void pack_f(const float* __restrict__ feat, bf16* __restrict__ fT) {
    __shared__ float tile[32][33];
    const int p0 = blockIdx.x * 32, c0 = blockIdx.y * 32;
    const int tx = threadIdx.x, ty = threadIdx.y;      // (32,8)
    #pragma unroll
    for (int i = 0; i < 4; ++i)
        tile[ty + 8*i][tx] = feat[(size_t)(c0 + ty + 8*i) * HWP + p0 + tx];
    __syncthreads();
    #pragma unroll
    for (int i = 0; i < 4; ++i)
        fT[(size_t)(p0 + ty + 8*i) * C_INC + c0 + tx] = (bf16)tile[tx][ty + 8*i];
}

// Repack conv weights OIHW f32 -> bf16 wtap[tap][co][ci]  ([N][K] per tap GEMM)
__global__ __launch_bounds__(256)
void pack_w(const float* __restrict__ conv_w, bf16* __restrict__ wtap) {
    const int idx = blockIdx.x * 256 + threadIdx.x;    // < 9*256*256
    const int ci  = idx & 255;
    const int co  = (idx >> 8) & 255;
    const int tap = idx >> 16;
    wtap[idx] = (bf16)conv_w[(size_t)(co * 256 + ci) * 9 + tap];
}

// Fold normalization + classifier + (regressor @ t_inv + mu) into one
// [80][256] bf16 weight matrix (cols 0..14 cls, 15..74 bbox, 75..79 pad) + bias.
__global__ __launch_bounds__(256)
void fold_head(const float* __restrict__ cls_w, const float* __restrict__ reg_w,
               const float* __restrict__ reg_b, const float* __restrict__ t_inv,
               const float* __restrict__ mu,    const float* __restrict__ stats_mean,
               const float* __restrict__ mean_norm,
               bf16* __restrict__ B2, float* __restrict__ bias2) {
    __shared__ float red[256];
    const int j = blockIdx.x, k = threadIdx.x;
    const float s = 20.0f / mean_norm[0];
    float wv = 0.0f;
    if (j < NCLS) {
        wv = cls_w[j * C_INC + k];
    } else if (j < NOUTC) {
        const int c = (j - NCLS) >> 2, h = (j - NCLS) & 3;
        #pragma unroll
        for (int g = 0; g < 4; ++g)
            wv += reg_w[(size_t)(c*4 + g) * C_INC + k] * t_inv[(c*4 + g)*4 + h];
    }
    wv *= s;
    B2[(size_t)j * C_INC + k] = (bf16)wv;
    red[k] = stats_mean[k] * wv;
    __syncthreads();
    for (int off = 128; off > 0; off >>= 1) {
        if (k < off) red[k] += red[k + off];
        __syncthreads();
    }
    if (k == 0) {
        float bb = 0.0f;
        if (j >= NCLS && j < NOUTC) {
            const int c = (j - NCLS) >> 2, h = (j - NCLS) & 3;
            #pragma unroll
            for (int g = 0; g < 4; ++g) bb += reg_b[c*4 + g] * t_inv[(c*4 + g)*4 + h];
            bb += mu[c*4 + h];
        }
        bias2[j] = bb - red[0];
    }
}

// ---------------------------------------------------------------------------
// Stage A: 3x3 conv as 9 shifted GEMMs.  128(M)x128(N) tile per WG, K=9*256.
// - A tile: global->LDS via async-DMA (ASYNCcnt), double-buffered (s+1 overlaps s).
// - B tile: 64KB per tap DMA'd by the Tensor Data Mover (tensor_load_to_lds,
//   TENSORcnt), ALSO double-buffered: tap t+1's DMA is issued at the top of
//   tap t and waited with s_wait_tensorcnt<=1 (in-order TDM completion), so
//   the 64KB transfer hides under the 512-WMMA tap body.  TDM padding (16B per
//   512B row) keeps ds_load_b128 fragment reads aligned and conflict-free.
__global__ __launch_bounds__(256)
void conv_gemm(const bf16* __restrict__ fT, const bf16* __restrict__ wtap,
               const float* __restrict__ conv_b, bf16* __restrict__ f0) {
    __shared__ bf16 As[2][128 * 40] __attribute__((aligned(16)));  //  20,480 B
    __shared__ bf16 Bs[2][128 * BROW] __attribute__((aligned(16))); // 135,168 B
    const int tid   = threadIdx.x;
    const int wave  = tid >> 5;
    const int lane  = tid & 31;
    const int lhalf = lane >> 4;        // 0/1
    const int lmod  = lane & 15;
    const int tile_m0 = blockIdx.x * 128;
    const int tile_n0 = blockIdx.y * 128;

    // cooperative loader: thread -> (row, 16-element half of the 32-K chunk)
    const int lrow = tid >> 1;
    const int lk   = (tid & 1) * 16;
    const int p_l  = tile_m0 + lrow;
    const int y_l  = p_l / WWD;
    const int x_l  = p_l - y_l * WWD;

    v8f acc[8];
    #pragma unroll
    for (int j = 0; j < 8; ++j)
        #pragma unroll
        for (int r = 0; r < 8; ++r) acc[j][r] = 0.0f;

    // stage K-step `st` of the A tile into LDS buffer `buf` (async DMA)
    auto stage = [&](int st, int buf) {
        const int tap = st >> 3;
        const int kin = (st & 7) * 32;
        const int dy  = tap / 3 - 1;
        const int dx  = tap % 3 - 1;
        bf16* dst = &As[buf][lrow * 40 + lk];
        if ((unsigned)(y_l + dy) < (unsigned)HH &&
            (unsigned)(x_l + dx) < (unsigned)WWD) {
            const unsigned ldsa = (unsigned)(size_t)dst;   // wave-relative LDS addr
            const bf16* src = fT + (size_t)(p_l + dy * WWD + dx) * C_INC + kin + lk;
            asm volatile("global_load_async_to_lds_b128 %0, %1, off"
                         :: "v"(ldsa), "v"(src) : "memory");
            asm volatile("global_load_async_to_lds_b128 %0, %1, off offset:16"
                         :: "v"(ldsa), "v"(src) : "memory");
        } else {
            const uint4 z = make_uint4(0u, 0u, 0u, 0u);    // conv halo: zero-fill
            *(uint4*)(dst)     = z;
            *(uint4*)(dst + 8) = z;
        }
    };

    // TDM: DMA tap's 128x256 bf16 B tile into Bs[buf] with 16B/row padding
    auto tdm_load_tap = [&](int tap, int buf) {
        const uint64_t ga  = (uint64_t)(size_t)(wtap + ((size_t)tap * C_INC + tile_n0) * C_INC);
        const uint32_t lds = (uint32_t)(size_t)(&Bs[buf][0]);
        u32x4 g0;
        g0[0] = 1u;                                        // count=1 (user mode)
        g0[1] = lds;                                       // lds_addr
        g0[2] = (uint32_t)ga;                              // global_addr[31:0]
        g0[3] = (uint32_t)((ga >> 32) & 0x01FFFFFFu) | (2u << 30);  // ga[56:32] | type=2
        u32x8 g1;
        g1[0] = (1u << 16) | (1u << 20) | (6u << 22) | (3u << 25);
                // data_size=1 (2B) | pad_enable | pad_interval=6 (512B) | pad_amount=3 (16B)
        g1[1] = 256u << 16;                                // tensor_dim0 = 256 (K)
        g1[2] = 128u << 16;                                // tensor_dim1 = 128 (rows)
        g1[3] = 256u << 16;                                // tile_dim0 = 256
        g1[4] = 128u;                                      // tile_dim1 = 128
        g1[5] = 256u;                                      // tensor_dim0_stride = 256
        g1[6] = 0u;
        g1[7] = 0u;
        asm volatile("tensor_load_to_lds %0, %1" :: "s"(g0), "s"(g1) : "memory");
    };

    stage(0, 0);
    if (wave == 0) tdm_load_tap(0, 0);

    for (int tap = 0; tap < 9; ++tap) {
        const int bcur = tap & 1;
        if (tap + 1 < 9) {
            // issue next tap's DMA into the other buffer (its readers finished
            // at tap-1's last inner barrier), then wait for the CURRENT tile
            // only: TDM ops of one wave complete in order -> tensorcnt<=1.
            if (wave == 0) tdm_load_tap(tap + 1, bcur ^ 1);
            wait_tensor1();            // no-op for non-issuing waves
        } else {
            wait_tensor0();
        }
        __syncthreads();               // Bs[bcur] holds this tap's weights

        #pragma unroll 1
        for (int k8 = 0; k8 < 8; ++k8) {
            const int step = tap * 8 + k8;
            const int cur  = step & 1;
            wait_async0();             // our slice of As[cur] landed
            __syncthreads();           // everyone's slice landed
            if (step + 1 < 72) stage(step + 1, cur ^ 1);

            const int kin = k8 * 32;

            // A fragment (16x32 bf16): two contiguous 16B runs per lane
            const bf16* ap = &As[cur][(wave * 16 + lmod) * 40 + lhalf * 8];
            const v8bf alo = *(const v8bf*)(ap);
            const v8bf ahi = *(const v8bf*)(ap + 16);
            const v16bf afrag = __builtin_shufflevector(alo, ahi,
                0,1,2,3,4,5,6,7,8,9,10,11,12,13,14,15);

            // B fragments from LDS: one base, imm offsets (j-stride 16*528B)
            const bf16* bbase = &Bs[bcur][lmod * BROW + (kin + lhalf * 8)];
            v16bf bfr[8];
            #pragma unroll
            for (int j = 0; j < 8; ++j) {
                const v8bf blo = *(const v8bf*)(bbase + j * 16 * BROW);
                const v8bf bhi = *(const v8bf*)(bbase + j * 16 * BROW + 16);
                bfr[j] = __builtin_shufflevector(blo, bhi,
                    0,1,2,3,4,5,6,7,8,9,10,11,12,13,14,15);
            }
            #pragma unroll
            for (int j = 0; j < 8; ++j)
                acc[j] = __builtin_amdgcn_wmma_f32_16x16x32_bf16(
                    false, afrag, false, bfr[j], (short)0, acc[j], false, false);
            __syncthreads();           // readers done before A-buffer reuse
        }
    }

    // epilogue: +bias, ReLU, store bf16 channels-last (A-ready for stage B)
    #pragma unroll
    for (int j = 0; j < 8; ++j) {
        const int co = tile_n0 + j * 16 + lmod;
        const float bb = conv_b[co];
        #pragma unroll
        for (int r = 0; r < 8; ++r) {
            const int p = tile_m0 + wave * 16 + r + (lhalf ? 8 : 0);
            float v = acc[j][r] + bb;
            v = v > 0.0f ? v : 0.0f;
            f0[(size_t)p * C_INC + co] = (bf16)v;
        }
    }
}

// ---------------------------------------------------------------------------
// Stage B: [60800x256] x [256x80] GEMM; epilogue transposed through LDS so the
// plane-major (logits/bbox) stores are contiguous.
__global__ __launch_bounds__(256)
void head_gemm(const bf16* __restrict__ f0, const bf16* __restrict__ B2,
               const float* __restrict__ bias2, float* __restrict__ out) {
    __shared__ float outs[128][84];
    const int tid   = threadIdx.x;
    const int wave  = tid >> 5;
    const int lane  = tid & 31;
    const int lhalf = lane >> 4;
    const int lmod  = lane & 15;
    const int m0    = blockIdx.x * 128 + wave * 16;

    v8f acc[5];
    #pragma unroll
    for (int j = 0; j < 5; ++j)
        #pragma unroll
        for (int r = 0; r < 8; ++r) acc[j][r] = 0.0f;

    const bf16* abase = f0 + (size_t)(m0 + lmod) * C_INC + lhalf * 16;
    const bf16* bcol  = B2 + (size_t)lmod * C_INC + lhalf * 16;

    #pragma unroll
    for (int ks = 0; ks < 8; ++ks) {
        const int kbase = ks * 32;
        const v8bf alo = *(const v8bf*)(abase + kbase);
        const v8bf ahi = *(const v8bf*)(abase + kbase + 8);
        const v16bf afrag = __builtin_shufflevector(alo, ahi,
            0,1,2,3,4,5,6,7,8,9,10,11,12,13,14,15);
        v16bf bfr[5];
        #pragma unroll
        for (int j = 0; j < 5; ++j)
            bfr[j] = *(const v16bf*)(bcol + kbase + (size_t)j * 16 * C_INC);
        #pragma unroll
        for (int j = 0; j < 5; ++j)
            acc[j] = __builtin_amdgcn_wmma_f32_16x16x32_bf16(
                false, afrag, false, bfr[j], (short)0, acc[j], false, false);
    }

    #pragma unroll
    for (int j = 0; j < 5; ++j)
        #pragma unroll
        for (int r = 0; r < 8; ++r)
            outs[wave * 16 + r + (lhalf ? 8 : 0)][j * 16 + lmod] = acc[j][r];
    __syncthreads();

    const int pbase = blockIdx.x * 128;
    for (int idx = tid; idx < NOUTC * 128; idx += 256) {
        const int c   = idx >> 7;
        const int row = idx & 127;
        out[(size_t)c * HWP + pbase + row] = outs[row][c] + bias2[c];
    }
}

// ---------------------------------------------------------------------------
extern "C" void kernel_launch(void* const* d_in, const int* in_sizes, int n_in,
                              void* d_out, int out_size, void* d_ws, size_t ws_size,
                              hipStream_t stream) {
    const float* feature    = (const float*)d_in[0];
    const float* conv_w     = (const float*)d_in[1];
    const float* conv_b     = (const float*)d_in[2];
    const float* stats_mean = (const float*)d_in[3];
    const float* mean_norm  = (const float*)d_in[4];
    const float* cls_w      = (const float*)d_in[5];
    const float* reg_w      = (const float*)d_in[6];
    const float* reg_b      = (const float*)d_in[7];
    const float* t_inv      = (const float*)d_in[8];
    const float* mu         = (const float*)d_in[9];

    char* ws = (char*)d_ws;
    bf16*  fT    = (bf16*)(ws + OFF_FT);
    bf16*  f0    = (bf16*)(ws + OFF_F0);
    bf16*  wtap  = (bf16*)(ws + OFF_WT);
    bf16*  B2    = (bf16*)(ws + OFF_B2);
    float* bias2 = (float*)(ws + OFF_BIAS);

    pack_f<<<dim3(HWP / 32, C_INC / 32), dim3(32, 8), 0, stream>>>(feature, fT);
    pack_w<<<(9 * C_INC * C_INC) / 256, 256, 0, stream>>>(conv_w, wtap);
    fold_head<<<NPADC, 256, 0, stream>>>(cls_w, reg_w, reg_b, t_inv, mu,
                                         stats_mean, mean_norm, B2, bias2);
    conv_gemm<<<dim3(HWP / 128, 2), 256, 0, stream>>>(fT, wtap, conv_b, f0);
    head_gemm<<<HWP / 128, 256, 0, stream>>>(f0, B2, bias2, (float*)d_out);
}
